// SchNetNet_80865644249575
// MI455X (gfx1250) — compile-verified, matrix operand order checked by reading, not run on hardware
//
#include <hip/hip_runtime.h>
#include <hip/hip_bf16.h>

typedef __attribute__((ext_vector_type(16))) _Float16 v16h;
typedef __attribute__((ext_vector_type(8)))  float    v8f;
typedef __attribute__((ext_vector_type(4)))  unsigned int v4u;
typedef __attribute__((ext_vector_type(8)))  int      v8i;
typedef __attribute__((ext_vector_type(4)))  int      v4i;

#define HIDDEN  128
#define FILTERS 128
#define NGAUSS  50
#define CUTOFFR 10.0f
#define WPB 4            // waves per block (block = 128 threads = 4 wave32)

// K-offset inside a 16x32 f16 A/B fragment for VGPR-slot j, half-wave hi.
__device__ __host__ inline int kofs(int j, int hi) {
  return ((j & 4) << 2) + (hi << 3) + ((j & 3) << 1);
}

__device__ inline float sspf(float x) {             // softplus(x) - ln 2
  float sp = (x > 20.0f) ? x : log1pf(__expf(x));
  return sp - 0.6931471805599453f;
}

__device__ inline v8f wmma16x16x32(v16h a, v16h b, v8f c) {
  return __builtin_amdgcn_wmma_f32_16x16x32_f16(false, a, false, b,
                                                (short)0, c, false, false);
}

// ---- TDM: DMA `nbytes` (multiple of 8) from global `src` into LDS offset
// `lds_off` as a 1-D tile of 8-byte elements.  D# layout per CDNA5 ISA ch.8.
__device__ inline void tdm_load_to_lds(unsigned lds_off, const void* src,
                                       unsigned nbytes) {
  unsigned ne = nbytes >> 3;                       // 8-byte elements
  unsigned long long ga = (unsigned long long)src;
  v4u g0 = { 1u,                                   // count=1, user descriptor
             lds_off,                              // lds_addr [63:32]
             (unsigned)ga,                         // global_addr lo
             ((unsigned)(ga >> 32) & 0x1FFFFFFu) | (2u << 30) }; // hi + type=2
  v8i g1 = { (int)(3u << 16),                      // data_size = 8B
             (int)((ne & 0xFFFFu) << 16),          // tensor_dim0[15:0]
             (int)(((ne >> 16) & 0xFFFFu) | (1u << 16)), // dim0 hi | tensor_dim1=1
             (int)(ne << 16),                      // tile_dim0
             0,                                    // tile_dim1=0 (1-D), tile_dim2=0
             (int)ne,                              // tensor_dim0_stride lo
             0, 0 };
  v4i z4 = {0, 0, 0, 0};
#if defined(__clang_major__) && (__clang_major__ >= 23)
  v8i z8 = {0, 0, 0, 0, 0, 0, 0, 0};
  __builtin_amdgcn_tensor_load_to_lds(g0, g1, z4, z4, z8, 0);
#else
  __builtin_amdgcn_tensor_load_to_lds(g0, g1, z4, z4, 0);
#endif
}

__device__ inline unsigned lds_offset_of(const void* p) {
  return (unsigned)(unsigned long long)
      (__attribute__((address_space(3))) const void*)p;
}

// ---------------------------------------------------------------- embedding
__global__ void embed_kernel(const float* __restrict__ x,
                             const float* __restrict__ emb,
                             float* __restrict__ h, int N) {
  int idx = blockIdx.x * blockDim.x + threadIdx.x;
  if (idx >= N * HIDDEN) return;
  int n = idx >> 7, f = idx & 127;
  int z = (int)(118.0f * x[n]);
  h[idx] = emb[z * HIDDEN + f];
}

// ------------------------------------------------------- per-edge d, cutoff
__global__ void edge_prep_kernel(const float* __restrict__ pos,
                                 const int* __restrict__ ei,
                                 float* __restrict__ d_e,
                                 float* __restrict__ c_e, int E) {
  int e = blockIdx.x * blockDim.x + threadIdx.x;
  if (e >= E) return;
  int s = ei[e], d = ei[E + e];
  float dx = pos[d*3+0] - pos[s*3+0];
  float dy = pos[d*3+1] - pos[s*3+1];
  float dz = pos[d*3+2] - pos[s*3+2];
  float dist = sqrtf(dx*dx + dy*dy + dz*dz);
  d_e[e] = dist;
  c_e[e] = 0.5f * (cosf(dist * 3.14159265358979f / CUTOFFR) + 1.0f);
}

// ------------------------- pack f32 weight (Krows x Ncols) into B-fragments
__global__ void pack_w_kernel(const float* __restrict__ src, _Float16* __restrict__ dst,
                              int Krows, int Ncols, int NT, int total) {
  int p = blockIdx.x * blockDim.x + threadIdx.x;
  if (p >= total) return;
  int t = p & 1, j = (p >> 1) & 7, lane = (p >> 4) & 31;
  int q = p >> 9;
  int nt = q % NT, ks = q / NT;
  int k = ks * 32 + kofs(j, lane >> 4) + t;
  int col = nt * 16 + (lane & 15);
  float v = (k < Krows) ? src[(size_t)k * Ncols + col] : 0.0f;
  dst[p] = (_Float16)v;
}

// ------------------------------- generic node GEMM: Out = [ssp](A@B + bias)
template<int KS, int NT, bool SSP, bool HASB>
__global__ void gemm_node_kernel(const float* __restrict__ A,
                                 const _Float16* __restrict__ Bp,
                                 const float* __restrict__ bias,
                                 float* __restrict__ Out,
                                 int Nrows, int ldA, int ldOut) {
  int lane = threadIdx.x & 31;
  int wave = threadIdx.x >> 5;
  int tile = blockIdx.x * (blockDim.x >> 5) + wave;
  int row0 = tile * 16;
  if (row0 >= Nrows) return;
  int m = lane & 15, hi = lane >> 4;
  int arow = row0 + m; if (arow >= Nrows) arow = Nrows - 1;
  const v16h* B = (const v16h*)Bp;
  v8f acc[NT];
#pragma unroll
  for (int nt = 0; nt < NT; ++nt) { v8f z = {}; acc[nt] = z; }
#pragma unroll
  for (int ks = 0; ks < KS; ++ks) {
    v16h a = {};
#pragma unroll
    for (int j = 0; j < 8; ++j) {
      int k0 = ks * 32 + kofs(j, hi);
      a[2*j]   = (_Float16)A[(size_t)arow * ldA + k0];
      a[2*j+1] = (_Float16)A[(size_t)arow * ldA + k0 + 1];
    }
#pragma unroll
    for (int nt = 0; nt < NT; ++nt) {
      v16h b = B[(ks * NT + nt) * 32 + lane];
      acc[nt] = wmma16x16x32(a, b, acc[nt]);
    }
  }
#pragma unroll
  for (int nt = 0; nt < NT; ++nt) {
    int f = nt * 16 + (lane & 15);
    float bv = HASB ? bias[f] : 0.0f;
#pragma unroll
    for (int r = 0; r < 8; ++r) {
      int row = row0 + r + 8 * hi;
      if (row < Nrows) {
        float v = acc[nt][r] + bv;
        if (SSP) v = sspf(v);
        Out[(size_t)row * ldOut + f] = v;
      }
    }
  }
}

// --- fused edge pipeline: gauss -> GEMM1 -> ssp -> GEMM2 -> *C -> gather/scatter
// W2 (32KB of packed B-fragments) is DMA'd into LDS once per block by the TDM.
__global__ void edge_msg_kernel(const float* __restrict__ d_e,
                                const float* __restrict__ c_e,
                                const int* __restrict__ ei,
                                const _Float16* __restrict__ w1p,
                                const _Float16* __restrict__ w2p,
                                const float* __restrict__ b1,
                                const float* __restrict__ b2,
                                const float* __restrict__ tbuf,
                                float* __restrict__ agg, int E) {
  __shared__ _Float16 w2lds[FILTERS * FILTERS];      // 32 KB packed W2
  __shared__ _Float16 stage[WPB][16 * FILTERS];      // 16 KB per-wave staging
  int lane = threadIdx.x & 31;
  int wave = threadIdx.x >> 5;
  int tile = blockIdx.x * WPB + wave;
  int e0 = tile * 16;

  if (wave == 0) {                                   // TDM issue: one wave
    tdm_load_to_lds(lds_offset_of(&w2lds[0]), w2p,
                    FILTERS * FILTERS * sizeof(_Float16));
    __builtin_amdgcn_s_wait_tensorcnt(0);            // TENSORcnt -> 0
  }
  __syncthreads();                                   // all waves see W2 in LDS
  if (e0 >= E) return;

  int m = lane & 15, hi = lane >> 4;
  __builtin_prefetch(w1p + (size_t)lane * 128, 0, 0); // global_prefetch_b8
  const float step  = CUTOFFR / 49.0f;
  const float coeff = -0.5f / (step * step);
  float dm = (e0 + m < E) ? d_e[e0 + m] : 1e30f;     // OOR rows -> attr 0
  const v16h* B1 = (const v16h*)w1p;
  const v16h* B2 = (const v16h*)w2lds;

  v8f acc[8];
#pragma unroll
  for (int nt = 0; nt < 8; ++nt) { v8f z = {}; acc[nt] = z; }
#pragma unroll
  for (int ks = 0; ks < 2; ++ks) {                   // K = 64 (50 padded)
    v16h a = {};
#pragma unroll
    for (int j = 0; j < 8; ++j) {
      int k0 = ks * 32 + kofs(j, hi);
      float g0 = 0.0f, g1 = 0.0f;
      if (k0 < NGAUSS)     { float u = dm - k0 * step;     g0 = __expf(coeff*u*u); }
      if (k0 + 1 < NGAUSS) { float u = dm - (k0+1) * step; g1 = __expf(coeff*u*u); }
      a[2*j] = (_Float16)g0; a[2*j+1] = (_Float16)g1;
    }
#pragma unroll
    for (int nt = 0; nt < 8; ++nt) {
      v16h b = B1[(ks * 8 + nt) * 32 + lane];
      acc[nt] = wmma16x16x32(a, b, acc[nt]);
    }
  }
  // bias + shifted softplus -> per-wave LDS staging (16x128 f16)
#pragma unroll
  for (int nt = 0; nt < 8; ++nt) {
    int f = nt * 16 + (lane & 15);
    float bv = b1[f];
#pragma unroll
    for (int r = 0; r < 8; ++r) {
      int row = r + 8 * hi;
      stage[wave][row * FILTERS + f] = (_Float16)sspf(acc[nt][r] + bv);
    }
  }
  asm volatile("s_wait_dscnt 0x0" ::: "memory");     // DS store->load fence

  v8f acc2[8];
#pragma unroll
  for (int nt = 0; nt < 8; ++nt) { v8f z = {}; acc2[nt] = z; }
#pragma unroll
  for (int ks = 0; ks < 4; ++ks) {                   // K = 128
    v16h a = {};
#pragma unroll
    for (int j = 0; j < 8; ++j) {
      int k0 = ks * 32 + kofs(j, hi);
      a[2*j]   = stage[wave][m * FILTERS + k0];
      a[2*j+1] = stage[wave][m * FILTERS + k0 + 1];
    }
#pragma unroll
    for (int nt = 0; nt < 8; ++nt) {
      v16h b = B2[(ks * 8 + nt) * 32 + lane];        // ds_load from TDM'd LDS
      acc2[nt] = wmma16x16x32(a, b, acc2[nt]);
    }
  }
  int se[8], de[8], ok[8]; float ce[8];
#pragma unroll
  for (int r = 0; r < 8; ++r) {
    int e = e0 + r + 8 * hi;
    ok[r] = (e < E);
    se[r] = ok[r] ? ei[e] : 0;
    de[r] = ok[r] ? ei[E + e] : 0;
    ce[r] = ok[r] ? c_e[e] : 0.0f;
  }
#pragma unroll
  for (int nt = 0; nt < 8; ++nt) {
    int f = nt * 16 + (lane & 15);
    float bv = b2[f];
#pragma unroll
    for (int r = 0; r < 8; ++r) {
      if (ok[r]) {
        float w  = (acc2[nt][r] + bv) * ce[r];
        float mv = tbuf[(size_t)se[r] * FILTERS + f] * w;
        atomicAdd(&agg[(size_t)de[r] * FILTERS + f], mv);
      }
    }
  }
}

// ----- fused node update: h += (ssp(agg@lin2 + b2)) @ lin + b
__global__ void node_update_kernel(const float* __restrict__ agg,
                                   const _Float16* __restrict__ l2p,
                                   const float* __restrict__ l2b,
                                   const _Float16* __restrict__ lwp,
                                   const float* __restrict__ lwb,
                                   float* __restrict__ h, int Nrows) {
  __shared__ _Float16 stage[WPB][16 * HIDDEN];
  int lane = threadIdx.x & 31;
  int wave = threadIdx.x >> 5;
  int tile = blockIdx.x * WPB + wave;
  int row0 = tile * 16;
  if (row0 >= Nrows) return;
  int m = lane & 15, hi = lane >> 4;
  int arow = row0 + m; if (arow >= Nrows) arow = Nrows - 1;
  const v16h* B1 = (const v16h*)l2p;
  const v16h* B2 = (const v16h*)lwp;

  v8f acc[8];
#pragma unroll
  for (int nt = 0; nt < 8; ++nt) { v8f z = {}; acc[nt] = z; }
#pragma unroll
  for (int ks = 0; ks < 4; ++ks) {
    v16h a = {};
#pragma unroll
    for (int j = 0; j < 8; ++j) {
      int k0 = ks * 32 + kofs(j, hi);
      a[2*j]   = (_Float16)agg[(size_t)arow * HIDDEN + k0];
      a[2*j+1] = (_Float16)agg[(size_t)arow * HIDDEN + k0 + 1];
    }
#pragma unroll
    for (int nt = 0; nt < 8; ++nt) {
      v16h b = B1[(ks * 8 + nt) * 32 + lane];
      acc[nt] = wmma16x16x32(a, b, acc[nt]);
    }
  }
#pragma unroll
  for (int nt = 0; nt < 8; ++nt) {
    int f = nt * 16 + (lane & 15);
    float bv = l2b[f];
#pragma unroll
    for (int r = 0; r < 8; ++r) {
      int row = r + 8 * hi;
      stage[wave][row * HIDDEN + f] = (_Float16)sspf(acc[nt][r] + bv);
    }
  }
  asm volatile("s_wait_dscnt 0x0" ::: "memory");

  v8f acc2[8];
#pragma unroll
  for (int nt = 0; nt < 8; ++nt) { v8f z = {}; acc2[nt] = z; }
#pragma unroll
  for (int ks = 0; ks < 4; ++ks) {
    v16h a = {};
#pragma unroll
    for (int j = 0; j < 8; ++j) {
      int k0 = ks * 32 + kofs(j, hi);
      a[2*j]   = stage[wave][m * HIDDEN + k0];
      a[2*j+1] = stage[wave][m * HIDDEN + k0 + 1];
    }
#pragma unroll
    for (int nt = 0; nt < 8; ++nt) {
      v16h b = B2[(ks * 8 + nt) * 32 + lane];
      acc2[nt] = wmma16x16x32(a, b, acc2[nt]);
    }
  }
#pragma unroll
  for (int nt = 0; nt < 8; ++nt) {
    int f = nt * 16 + (lane & 15);
    float bv = lwb[f];
#pragma unroll
    for (int r = 0; r < 8; ++r) {
      int row = row0 + r + 8 * hi;
      if (row < Nrows)
        h[(size_t)row * HIDDEN + f] += acc2[nt][r] + bv;
    }
  }
}

// -------------------------- final head: s = h1@out2_w + b ; per-graph sum
__global__ void out_final_kernel(const float* __restrict__ h1,
                                 const float* __restrict__ w2,
                                 const float* __restrict__ b2,
                                 const int* __restrict__ batch,
                                 float* __restrict__ out, int N) {
  int nd = blockIdx.x * blockDim.x + threadIdx.x;
  if (nd >= N) return;
  float s = b2[0];
#pragma unroll 8
  for (int f = 0; f < 64; ++f) s += h1[(size_t)nd * 64 + f] * w2[f];
  atomicAdd(&out[batch[nd]], s);
}

extern "C" void kernel_launch(void* const* d_in, const int* in_sizes, int n_in,
                              void* d_out, int out_size, void* d_ws, size_t ws_size,
                              hipStream_t stream) {
  (void)n_in; (void)ws_size;
  const float* x      = (const float*)d_in[0];
  const float* pos    = (const float*)d_in[1];
  const float* emb    = (const float*)d_in[2];
  const float* mlp_w1 = (const float*)d_in[3];
  const float* mlp_b1 = (const float*)d_in[4];
  const float* mlp_w2 = (const float*)d_in[5];
  const float* mlp_b2 = (const float*)d_in[6];
  const float* lin1_w = (const float*)d_in[7];
  const float* lin2_w = (const float*)d_in[8];
  const float* lin2_b = (const float*)d_in[9];
  const float* lin_w  = (const float*)d_in[10];
  const float* lin_b  = (const float*)d_in[11];
  const float* out1_w = (const float*)d_in[12];
  const float* out1_b = (const float*)d_in[13];
  const float* out2_w = (const float*)d_in[14];
  const float* out2_b = (const float*)d_in[15];
  const int*   ei     = (const int*)d_in[16];
  const int*   batch  = (const int*)d_in[17];
  float* out = (float*)d_out;

  const int N = in_sizes[0];          // x is (N,1)
  const int E = in_sizes[16] / 2;     // edge_index is (2,E)

  char* ws = (char*)d_ws;
  size_t off = 0;
  auto carve = [&](size_t bytes) -> char* {
    char* p = ws + off;
    off = (off + bytes + 255) & ~((size_t)255);
    return p;
  };
  float*    h    = (float*)carve((size_t)N * 128 * 4);
  float*    tbuf = (float*)carve((size_t)N * 128 * 4);
  float*    agg  = (float*)carve((size_t)N * 128 * 4);
  float*    h1   = (float*)carve((size_t)N * 64 * 4);
  float*    d_e  = (float*)carve((size_t)E * 4);
  float*    c_e  = (float*)carve((size_t)E * 4);
  _Float16* w1p  = (_Float16*)carve((size_t)3 * 64  * 128 * 2);
  _Float16* w2p  = (_Float16*)carve((size_t)3 * 128 * 128 * 2);
  _Float16* l1p  = (_Float16*)carve((size_t)3 * 128 * 128 * 2);
  _Float16* l2p  = (_Float16*)carve((size_t)3 * 128 * 128 * 2);
  _Float16* lwp  = (_Float16*)carve((size_t)3 * 128 * 128 * 2);
  _Float16* o1p  = (_Float16*)carve((size_t)128 * 64 * 2);

  hipMemsetAsync(out, 0, (size_t)out_size * sizeof(float), stream);

  embed_kernel<<<(N * 128 + 255) / 256, 256, 0, stream>>>(x, emb, h, N);
  edge_prep_kernel<<<(E + 255) / 256, 256, 0, stream>>>(pos, ei, d_e, c_e, E);

  for (int i = 0; i < 3; ++i) {
    pack_w_kernel<<<(2*8*512 + 255)/256, 256, 0, stream>>>(
        mlp_w1 + (size_t)i*50*128,  w1p + (size_t)i*64*128,  50, 128, 8, 2*8*512);
    pack_w_kernel<<<(4*8*512 + 255)/256, 256, 0, stream>>>(
        mlp_w2 + (size_t)i*128*128, w2p + (size_t)i*128*128, 128,128, 8, 4*8*512);
    pack_w_kernel<<<(4*8*512 + 255)/256, 256, 0, stream>>>(
        lin1_w + (size_t)i*128*128, l1p + (size_t)i*128*128, 128,128, 8, 4*8*512);
    pack_w_kernel<<<(4*8*512 + 255)/256, 256, 0, stream>>>(
        lin2_w + (size_t)i*128*128, l2p + (size_t)i*128*128, 128,128, 8, 4*8*512);
    pack_w_kernel<<<(4*8*512 + 255)/256, 256, 0, stream>>>(
        lin_w  + (size_t)i*128*128, lwp + (size_t)i*128*128, 128,128, 8, 4*8*512);
  }
  pack_w_kernel<<<(4*4*512 + 255)/256, 256, 0, stream>>>(out1_w, o1p, 128, 64, 4, 4*4*512);

  const int node_tiles  = (N + 15) / 16;
  const int node_blocks = (node_tiles + WPB - 1) / WPB;
  const int edge_tiles  = (E + 15) / 16;
  const int edge_blocks = (edge_tiles + WPB - 1) / WPB;

  for (int i = 0; i < 3; ++i) {
    gemm_node_kernel<4, 8, false, false><<<node_blocks, 32 * WPB, 0, stream>>>(
        h, l1p + (size_t)i*128*128, nullptr, tbuf, N, 128, 128);
    hipMemsetAsync(agg, 0, (size_t)N * 128 * 4, stream);
    edge_msg_kernel<<<edge_blocks, 32 * WPB, 0, stream>>>(
        d_e, c_e, ei,
        w1p + (size_t)i*64*128, w2p + (size_t)i*128*128,
        mlp_b1 + (size_t)i*128, mlp_b2 + (size_t)i*128,
        tbuf, agg, E);
    node_update_kernel<<<node_blocks, 32 * WPB, 0, stream>>>(
        agg, l2p + (size_t)i*128*128, lin2_b + (size_t)i*128,
        lwp + (size_t)i*128*128, lin_b + (size_t)i*128, h, N);
  }
  gemm_node_kernel<4, 4, true, true><<<node_blocks, 32 * WPB, 0, stream>>>(
      h, o1p, out1_b, h1, N, 128, 64);
  out_final_kernel<<<(N + 255) / 256, 256, 0, stream>>>(h1, out2_w, out2_b, batch, out, N);
}
